// MonotoneLayer_57586921505191
// MI455X (gfx1250) — compile-verified, matrix operand order checked by reading, not run on hardware
//
#include <hip/hip_runtime.h>
#include <math.h>

typedef __attribute__((ext_vector_type(16))) __bf16 v16bf;
typedef __attribute__((ext_vector_type(8)))  float  v8f;
typedef __attribute__((ext_vector_type(4)))  float  f32x4;
typedef __attribute__((ext_vector_type(4)))  unsigned int u32x4;
typedef __attribute__((ext_vector_type(8)))  int    i32x8;
typedef __attribute__((ext_vector_type(4)))  int    i32x4;

#define MDIM   32
#define MH1    8
#define NROWS  262144

// Packed region layout (same in d_ws and in LDS):
//   Aw  : bf16 [32 d][5 tile][32 lane][16 elem]   163840 @ 0
//   bmx : f32  [32 d][16]  exp(wlast[d,d,:]) pad    2048 @ 163840
//   blp : f32  [32 d][16]  blast[d,:] padded        2048 @ 165888
//   b0c : f32  [32 d][32]  bs0 copy                 4096 @ 167936
//   b1c : f32  [32 d][32]  bs1 copy                 4096 @ 172032
#define SM_AW    0
#define SM_BMX   163840
#define SM_BLP   165888
#define SM_B0    167936
#define SM_B1    172032
#define SM_BYTES 176128   // 22016 x 8B

__device__ __forceinline__ float relu1(float x) {   // single v_med3_f32
    return __builtin_amdgcn_fmed3f(x, 0.0f, 3.402823466e+38f);
}

// K index held by element e of a 16-bit A/B fragment for lane-half `half`.
__device__ __forceinline__ int frag_k(int e, int half) {
    return (e < 8) ? (half * 8 + e) : (16 + half * 8 + (e - 8));
}

// ---------------------------------------------------------------------------
// Prep: pack weights into WMMA A-fragment layout (transposed world) + bias
// tables, all into one contiguous region so the main kernel TDM-copies it.
// ---------------------------------------------------------------------------
__global__ __launch_bounds__(256) void mono_prep_kernel(
    const float* __restrict__ ws0, const float* __restrict__ bs0,
    const float* __restrict__ ws1, const float* __restrict__ bs1,
    const float* __restrict__ wlast, const float* __restrict__ blast,
    unsigned char* __restrict__ wsbase)
{
    const int idx = blockIdx.x * 256 + threadIdx.x;

    __bf16* Aw  = (__bf16*)(wsbase + SM_AW);
    float*  bmx = (float*)(wsbase + SM_BMX);
    float*  blp = (float*)(wsbase + SM_BLP);
    float*  b0c = (float*)(wsbase + SM_B0);
    float*  b1c = (float*)(wsbase + SM_B1);

    if (idx < 32 * 5 * 32) {
        const int lane = idx & 31;
        const int t    = (idx >> 5) % 5;
        const int d    = idx / (5 * 32);
        const int half = lane >> 4;
        const int jl   = lane & 15;
        __bf16* dst = Aw + (size_t)idx * 16;
#pragma unroll
        for (int e = 0; e < 16; ++e) {
            const int k = frag_k(e, half);
            float v;
            if (t < 2) {                       // W0^T, rows k>=d zeroed
                const int j = t * 16 + jl;
                v = (k < d) ? ws0[d * 1024 + k * 32 + j] : 0.0f;
            } else if (t < 4) {                // W1^T, no mask needed
                const int j = (t - 2) * 16 + jl;
                v = ws1[d * 1024 + k * 32 + j];
            } else {                           // Wmono^T, k>=d and j>=8 zeroed
                const int j = jl;
                v = (k < d && j < MH1) ? wlast[d * 256 + k * MH1 + j] : 0.0f;
            }
            dst[e] = (__bf16)v;
        }
    }

    if (idx < 32 * 16) {
        const int d = idx >> 4, j = idx & 15;
        bmx[idx] = (j < MH1) ? expf(wlast[d * 256 + d * MH1 + j]) : 0.0f;
        blp[idx] = (j < MH1) ? blast[d * MH1 + j] : 0.0f;
    }
    if (idx < 32 * 32) {
        b0c[idx] = bs0[idx];
        b1c[idx] = bs1[idx];
    }
}

// ---------------------------------------------------------------------------
// TDM: DMA the whole packed region (176128 B = 22016 x 8B, 1-D tile) into LDS.
// 6-arg builtin (clang-23 / therock HIP): (g0, g1, g2, g3, g4, cpol).
// ---------------------------------------------------------------------------
__device__ __forceinline__ void tdm_load_all(const void* gsrc, unsigned lds_off) {
    const unsigned long long ga = (unsigned long long)(uintptr_t)gsrc;
    const unsigned n64 = SM_BYTES / 8;         // 22016 elements of 8 bytes

    u32x4 g0;
    g0[0] = 1u;                                 // count=1 (valid user D#)
    g0[1] = lds_off;                            // lds_addr (bytes)
    g0[2] = (unsigned)(ga & 0xffffffffu);       // global_addr[31:0]
    g0[3] = (unsigned)((ga >> 32) & 0x01ffffffu) | (2u << 30);  // ga[56:32] | type=2

    i32x8 g1;
    g1[0] = (int)(3u << 16);                    // workgroup_mask=0, data_size=3 (8B)
    g1[1] = (int)((n64 & 0xffffu) << 16);       // tensor_dim0[15:0]
    g1[2] = (int)((n64 >> 16) & 0xffffu)        // tensor_dim0[31:16]
          | (1 << 16);                          // tensor_dim1 = 1
    g1[3] = (int)((n64 & 0xffffu) << 16);       // tile_dim0 = 22016
    g1[4] = 0;                                  // tile_dim1/2 = 0 (unused)
    g1[5] = (int)n64;                           // tensor_dim0_stride
    g1[6] = 0;
    g1[7] = 0;

    const i32x4 z4 = {0, 0, 0, 0};              // groups 2/3 unused (<=2D)
    const i32x8 z8 = {0, 0, 0, 0, 0, 0, 0, 0};
    __builtin_amdgcn_tensor_load_to_lds(g0, g1, z4, z4, z8, 0);
}

// ---------------------------------------------------------------------------
// Main: one wave = 32 rows (two 16-row sample tiles sharing every A-fragment
// and bias C-fragment); 16 waves/block. GEMMs as h^T = W^T @ x^T so each f32
// D-fragment is directly the next WMMA's bf16 B-fragment. Biases ride the
// WMMA C operand. All weights/biases served from LDS (TDM-staged).
// ---------------------------------------------------------------------------
__global__ __launch_bounds__(512) void mono_main_kernel(
    const float* __restrict__ x, const unsigned char* __restrict__ wsbase,
    float* __restrict__ out)
{
    __shared__ __align__(16) unsigned char smem[SM_BYTES];

    if (threadIdx.x < 32) {                     // wave 0 issues the TDM copy
        tdm_load_all(wsbase, (unsigned)(uintptr_t)&smem[0]);
        __builtin_amdgcn_s_wait_tensorcnt(0);
    }
    __syncthreads();

    const __bf16* smAw  = (const __bf16*)(smem + SM_AW);
    const float*  smBmx = (const float*)(smem + SM_BMX);
    const float*  smBlp = (const float*)(smem + SM_BLP);
    const float*  smB0  = (const float*)(smem + SM_B0);
    const float*  smB1  = (const float*)(smem + SM_B1);

    const int lane    = threadIdx.x & 31;
    const int wave    = threadIdx.x >> 5;
    const int rowbase = (blockIdx.x * 16 + wave) * 32;
    const int half    = lane >> 4;
    const int m16     = lane & 15;

    const float* xrA = x + (size_t)(rowbase + m16) * MDIM;
    const float* xrB = xrA + 16 * MDIM;

    // x^T B-fragments (K chunks: half*8..+7 and 16+half*8..+7).
    v16bf xB0, xB1;
    {
        const f32x4 a0 = *(const f32x4*)(xrA + half * 8);
        const f32x4 a1 = *(const f32x4*)(xrA + half * 8 + 4);
        const f32x4 a2 = *(const f32x4*)(xrA + 16 + half * 8);
        const f32x4 a3 = *(const f32x4*)(xrA + 16 + half * 8 + 4);
        const f32x4 b0 = *(const f32x4*)(xrB + half * 8);
        const f32x4 b1 = *(const f32x4*)(xrB + half * 8 + 4);
        const f32x4 b2 = *(const f32x4*)(xrB + 16 + half * 8);
        const f32x4 b3 = *(const f32x4*)(xrB + 16 + half * 8 + 4);
#pragma unroll
        for (int e = 0; e < 4; ++e) {
            xB0[e]      = (__bf16)a0[e];
            xB0[4 + e]  = (__bf16)a1[e];
            xB0[8 + e]  = (__bf16)a2[e];
            xB0[12 + e] = (__bf16)a3[e];
            xB1[e]      = (__bf16)b0[e];
            xB1[4 + e]  = (__bf16)b1[e];
            xB1[8 + e]  = (__bf16)b2[e];
            xB1[12 + e] = (__bf16)b3[e];
        }
    }

    const v8f vzero = {0.f, 0.f, 0.f, 0.f, 0.f, 0.f, 0.f, 0.f};

    for (int d = 0; d < MDIM; ++d) {
        const __bf16* af = smAw + ((size_t)(d * 5) * 32 + lane) * 16;

        // ---- GEMM1: h1^T = W0^T @ x^T + b0 (bias via C operand) ----
        const v16bf a0 = *(const v16bf*)(af);
        const v16bf a1 = *(const v16bf*)(af + 32 * 16);
        const v8f  cb00 = *(const v8f*)(smB0 + d * 32 + half * 8);
        const v8f  cb01 = *(const v8f*)(smB0 + d * 32 + 16 + half * 8);
        v8f h10a = __builtin_amdgcn_wmma_f32_16x16x32_bf16(
            false, a0, false, xB0, (short)0, cb00, false, false);
        v8f h11a = __builtin_amdgcn_wmma_f32_16x16x32_bf16(
            false, a1, false, xB0, (short)0, cb01, false, false);
        v8f h10b = __builtin_amdgcn_wmma_f32_16x16x32_bf16(
            false, a0, false, xB1, (short)0, cb00, false, false);
        v8f h11b = __builtin_amdgcn_wmma_f32_16x16x32_bf16(
            false, a1, false, xB1, (short)0, cb01, false, false);

        v16bf hBa, hBb;
#pragma unroll
        for (int r = 0; r < 8; ++r) {
            hBa[r]     = (__bf16)relu1(h10a[r]);
            hBa[8 + r] = (__bf16)relu1(h11a[r]);
            hBb[r]     = (__bf16)relu1(h10b[r]);
            hBb[8 + r] = (__bf16)relu1(h11b[r]);
        }

        // ---- GEMM2: h2^T = W1^T @ h1^T + b1 ----
        const v16bf a2 = *(const v16bf*)(af + 2 * 32 * 16);
        const v8f  cb10 = *(const v8f*)(smB1 + d * 32 + half * 8);
        v8f h20a = __builtin_amdgcn_wmma_f32_16x16x32_bf16(
            false, a2, false, hBa, (short)0, cb10, false, false);
        v8f h20b = __builtin_amdgcn_wmma_f32_16x16x32_bf16(
            false, a2, false, hBb, (short)0, cb10, false, false);
        v8f h21a = vzero, h21b = vzero;
        if (d >= 17) {   // wave-uniform: tile1's K-rows only alive when d>16
            const v16bf a3 = *(const v16bf*)(af + 3 * 32 * 16);
            const v8f cb11 = *(const v8f*)(smB1 + d * 32 + 16 + half * 8);
            h21a = __builtin_amdgcn_wmma_f32_16x16x32_bf16(
                false, a3, false, hBa, (short)0, cb11, false, false);
            h21b = __builtin_amdgcn_wmma_f32_16x16x32_bf16(
                false, a3, false, hBb, (short)0, cb11, false, false);
        }

        v16bf gBa, gBb;
#pragma unroll
        for (int r = 0; r < 8; ++r) {
            gBa[r]     = (__bf16)relu1(h20a[r]);
            gBa[8 + r] = (__bf16)relu1(h21a[r]);
            gBb[r]     = (__bf16)relu1(h20b[r]);
            gBb[8 + r] = (__bf16)relu1(h21b[r]);
        }

        // ---- GEMM3: xlayer^T = Wm^T @ h2^T + blast (C operand) ----
        const v16bf a4  = *(const v16bf*)(af + 4 * 32 * 16);
        const v8f  cbl  = *(const v8f*)(smBlp + d * 16 + half * 8);
        const v8f acca = __builtin_amdgcn_wmma_f32_16x16x32_bf16(
            false, a4, false, gBa, (short)0, cbl, false, false);
        const v8f accb = __builtin_amdgcn_wmma_f32_16x16x32_bf16(
            false, a4, false, gBb, (short)0, cbl, false, false);

        // ---- f32 epilogue: + x_d * exp(wlast[d,d,:]); min-pairs / max-groups ----
        const float* bm  = smBmx + d * 16 + half * 8;
        const float  xdA = xrA[d];
        const float  xdB = xrB[d];
        float va[8], vb[8];
#pragma unroll
        for (int r = 0; r < 8; ++r) {
            const float m = bm[r];
            va[r] = fmaf(xdA, m, acca[r]);
            vb[r] = fmaf(xdB, m, accb[r]);
        }
        const float resA = fmaxf(fmaxf(fminf(va[0], va[1]), fminf(va[2], va[3])),
                                 fmaxf(fminf(va[4], va[5]), fminf(va[6], va[7])));
        const float resB = fmaxf(fmaxf(fminf(vb[0], vb[1]), fminf(vb[2], vb[3])),
                                 fmaxf(fminf(vb[4], vb[5]), fminf(vb[6], vb[7])));

        if (lane < 16) {
            out[(size_t)(rowbase + m16) * MDIM + d]      = resA;
            out[(size_t)(rowbase + 16 + m16) * MDIM + d] = resB;
        }
    }
}

extern "C" void kernel_launch(void* const* d_in, const int* in_sizes, int n_in,
                              void* d_out, int out_size, void* d_ws, size_t ws_size,
                              hipStream_t stream) {
    const float* x     = (const float*)d_in[0];
    const float* ws0   = (const float*)d_in[1];
    const float* bs0   = (const float*)d_in[2];
    const float* ws1   = (const float*)d_in[3];
    const float* bs1   = (const float*)d_in[4];
    const float* wlast = (const float*)d_in[5];
    const float* blast = (const float*)d_in[6];

    unsigned char* wsbase = (unsigned char*)d_ws;

    // Pack weights (WMMA fragment layout) + bias tables into one region.
    mono_prep_kernel<<<20, 256, 0, stream>>>(ws0, bs0, ws1, bs1, wlast, blast,
                                             wsbase);

    // 32 rows/wave, 16 waves/block -> 512 rows/block.
    const int blocks = NROWS / 512;
    mono_main_kernel<<<blocks, 512, 0, stream>>>(x, wsbase, (float*)d_out);
}
